// InterCL_36670430773286
// MI455X (gfx1250) — matrix operand order repaired and checked by reading, not compiled, not run on hardware
//
#include <hip/hip_runtime.h>
#include <math.h>

#define BS 4096
#define DIM 1024

typedef __attribute__((ext_vector_type(16))) _Float16 v16h;
typedef __attribute__((ext_vector_type(8)))  _Float16 v8h;
typedef __attribute__((ext_vector_type(8)))  float    v8f;

// ---------------- 1. inverse L2 norms (fp32, eps = 1e-12) ----------------
__global__ void knorm(const float* __restrict__ img, const float* __restrict__ tab,
                      float* __restrict__ inv_img, float* __restrict__ inv_tab) {
    __shared__ float red[256];
    int b = blockIdx.x;                         // 0 .. 2*BS-1
    const float* src = (b < BS) ? img : tab;
    float*       dst = (b < BS) ? inv_img : inv_tab;
    int row = (b < BS) ? b : b - BS;
    const float* p = src + (size_t)row * DIM;
    float s = 0.f;
    for (int k = threadIdx.x; k < DIM; k += 256) { float v = p[k]; s += v * v; }
    red[threadIdx.x] = s; __syncthreads();
    for (int off = 128; off > 0; off >>= 1) {
        if (threadIdx.x < off) red[threadIdx.x] += red[threadIdx.x + off];
        __syncthreads();
    }
    if (threadIdx.x == 0) dst[row] = 1.0f / fmaxf(sqrtf(red[0]), 1e-12f);
}

// ---------------- 2. normalize + convert to f16 ----------------
__global__ void kconvert(const float* __restrict__ img, const float* __restrict__ tab,
                         const float* __restrict__ inv_img, const float* __restrict__ inv_tab,
                         _Float16* __restrict__ img_h, _Float16* __restrict__ tab_h) {
    size_t t = (size_t)blockIdx.x * 256 + threadIdx.x;   // one float4 per thread
    size_t per = (size_t)BS * DIM / 4;                   // 1M float4 per tensor
    const float* src = img; const float* inv = inv_img; _Float16* dst = img_h;
    size_t i = t;
    if (t >= per) { src = tab; inv = inv_tab; dst = tab_h; i = t - per; }
    size_t e = i * 4;
    int row = (int)(e >> 10);                            // DIM = 1024
    float s = inv[row];
    float4 v = ((const float4*)src)[i];
    dst[e + 0] = (_Float16)(v.x * s);
    dst[e + 1] = (_Float16)(v.y * s);
    dst[e + 2] = (_Float16)(v.z * s);
    dst[e + 3] = (_Float16)(v.w * s);
}

// ---------------- 3a. count of label==1 ----------------
__global__ void kcount(const int* __restrict__ lab, float* __restrict__ n1out) {
    __shared__ int red[256];
    int s = 0;
    for (int i = threadIdx.x; i < BS; i += 256) s += (lab[i] != 0) ? 1 : 0;
    red[threadIdx.x] = s; __syncthreads();
    for (int off = 128; off > 0; off >>= 1) {
        if (threadIdx.x < off) red[threadIdx.x] += red[threadIdx.x + off];
        __syncthreads();
    }
    if (threadIdx.x == 0) n1out[0] = (float)red[0];
}

// ---------------- 3b. per-row q / log q values ----------------
// Row i of label_ has c_i ones (same-class count); q row after softmax has only
// two distinct values: qs (same-class) and qd (different-class).
__global__ void kqprep(const int* __restrict__ lab, const float* __restrict__ n1p,
                       float* __restrict__ qs, float* __restrict__ qd,
                       float* __restrict__ lqs, float* __restrict__ lqd) {
    int i = blockIdx.x * 256 + threadIdx.x;
    if (i >= BS) return;
    float n1 = n1p[0];
    float c  = (lab[i] != 0) ? n1 : ((float)BS - n1);    // c >= 1 (includes self)
    float e1 = expf(1.0f / c);
    float Dv = c * e1 + ((float)BS - c);
    float qsv = e1 / Dv, qdv = 1.0f / Dv;
    qs[i] = qsv; qd[i] = qdv; lqs[i] = logf(qsv); lqd[i] = logf(qdv);
}

// ---------------- 4. WMMA GEMM: S = exp(scale) * img_n @ tab_n^T ----------------
// Wave computes a 32x32 tile (2x2 WMMA accumulators). Block = 8 waves -> 128x64.
__global__ void kgemm(const _Float16* __restrict__ Ah, const _Float16* __restrict__ Bh,
                      const float* __restrict__ scale, float* __restrict__ S) {
    const int lane = threadIdx.x & 31;
    const int wv   = threadIdx.x >> 5;
    const int hi   = lane >> 4;          // lane half (ISA fragment layout)
    const int l16  = lane & 15;
    const int m0 = blockIdx.x * 128 + (wv & 3) * 32;
    const int n0 = blockIdx.y * 64  + (wv >> 2) * 32;

    const _Float16* pa0 = Ah + (size_t)(m0 + l16) * DIM;
    const _Float16* pa1 = pa0 + (size_t)16 * DIM;
    const _Float16* pb0 = Bh + (size_t)(n0 + l16) * DIM;
    const _Float16* pb1 = pb0 + (size_t)16 * DIM;

    const int aoff = hi * 8;             // A: K = {aoff..aoff+7} and {aoff+16..aoff+23}
    const int boff = hi * 8;             // B: K = {boff..boff+7} and {boff+16..boff+23}

    v8f acc00 = {}, acc01 = {}, acc10 = {}, acc11 = {};
    for (int k = 0; k < DIM; k += 32) {
        v16h a0, a1, b0, b1;
        ((v8h*)&a0)[0] = *(const v8h*)(pa0 + k + aoff);
        ((v8h*)&a0)[1] = *(const v8h*)(pa0 + k + 16 + aoff);
        ((v8h*)&a1)[0] = *(const v8h*)(pa1 + k + aoff);
        ((v8h*)&a1)[1] = *(const v8h*)(pa1 + k + 16 + aoff);
        ((v8h*)&b0)[0] = *(const v8h*)(pb0 + k + boff);
        ((v8h*)&b0)[1] = *(const v8h*)(pb0 + k + 16 + boff);
        ((v8h*)&b1)[0] = *(const v8h*)(pb1 + k + boff);
        ((v8h*)&b1)[1] = *(const v8h*)(pb1 + k + 16 + boff);
        acc00 = __builtin_amdgcn_wmma_f32_16x16x32_f16(false, a0, false, b0, (short)0, acc00, false, false);
        acc01 = __builtin_amdgcn_wmma_f32_16x16x32_f16(false, a0, false, b1, (short)0, acc01, false, false);
        acc10 = __builtin_amdgcn_wmma_f32_16x16x32_f16(false, a1, false, b0, (short)0, acc10, false, false);
        acc11 = __builtin_amdgcn_wmma_f32_16x16x32_f16(false, a1, false, b1, (short)0, acc11, false, false);
    }
    float sexp = expf(scale[0]);
    const int mc = hi * 8;               // C/D layout: M = v + 8*hi, N = l16
#pragma unroll
    for (int v = 0; v < 8; ++v) {
        int m = m0 + mc + v;
        int n = n0 + l16;
        size_t r0 = (size_t)m * BS;
        size_t r1 = (size_t)(m + 16) * BS;
        S[r0 + n]      = sexp * acc00[v];
        S[r0 + n + 16] = sexp * acc01[v];
        S[r1 + n]      = sexp * acc10[v];
        S[r1 + n + 16] = sexp * acc11[v];
    }
}

// ---------------- 5a. row logsumexp ----------------
__global__ void krowlse(const float* __restrict__ S, float* __restrict__ rowLSE) {
    __shared__ float red[256];
    const float* p = S + (size_t)blockIdx.x * BS;
    float m = -3.4e38f;
    for (int j = threadIdx.x; j < BS; j += 256) m = fmaxf(m, p[j]);
    red[threadIdx.x] = m; __syncthreads();
    for (int off = 128; off > 0; off >>= 1) {
        if (threadIdx.x < off) red[threadIdx.x] = fmaxf(red[threadIdx.x], red[threadIdx.x + off]);
        __syncthreads();
    }
    m = red[0]; __syncthreads();
    float s = 0.f;
    for (int j = threadIdx.x; j < BS; j += 256) s += expf(p[j] - m);
    red[threadIdx.x] = s; __syncthreads();
    for (int off = 128; off > 0; off >>= 1) {
        if (threadIdx.x < off) red[threadIdx.x] += red[threadIdx.x + off];
        __syncthreads();
    }
    if (threadIdx.x == 0) rowLSE[blockIdx.x] = m + logf(red[0]);
}

// ---------------- 5b. column logsumexp (64 columns per block, coalesced) ----------------
__global__ void kcollse(const float* __restrict__ S, float* __restrict__ colLSE) {
    __shared__ float red[256];
    int c  = blockIdx.x * 64 + (threadIdx.x & 63);
    int rc = threadIdx.x >> 6;           // 0..3 row-chunks
    float m = -3.4e38f;
    for (int r = rc; r < BS; r += 4) m = fmaxf(m, S[(size_t)r * BS + c]);
    red[threadIdx.x] = m; __syncthreads();
    if (rc == 0) {
        int t = threadIdx.x;
        m = fmaxf(fmaxf(red[t], red[t + 64]), fmaxf(red[t + 128], red[t + 192]));
        red[t] = m;
    }
    __syncthreads();
    m = red[threadIdx.x & 63];
    __syncthreads();
    float s = 0.f;
    for (int r = rc; r < BS; r += 4) s += expf(S[(size_t)r * BS + c] - m);
    red[threadIdx.x] = s; __syncthreads();
    if (rc == 0) {
        int t = threadIdx.x;
        s = red[t] + red[t + 64] + red[t + 128] + red[t + 192];
        colLSE[c] = m + logf(s);
    }
}

// ---------------- 6. fused KL terms over S, per-block partial sums ----------------
__global__ void kreduce(const float* __restrict__ S, const float* __restrict__ rowLSE,
                        const float* __restrict__ colLSE, const int* __restrict__ lab,
                        const float* __restrict__ qs, const float* __restrict__ qd,
                        const float* __restrict__ lqs, const float* __restrict__ lqd,
                        float* __restrict__ partials) {
    __shared__ float red[256];
    size_t g = (size_t)blockIdx.x * 256 + threadIdx.x;   // 1M threads, 16 elems each
    int i  = (int)(g >> 8);
    int j0 = (int)(g & 255) * 16;
    int   li  = lab[i];
    float rl  = rowLSE[i];
    float qsi = qs[i], qdi = qd[i], lqsi = lqs[i], lqdi = lqd[i];
    const float* prow = S + (size_t)i * BS;
    float acc = 0.f;
#pragma unroll
    for (int jj = 0; jj < 16; ++jj) {
        int j = j0 + jj;
        float sij = prow[j];
        float lp1 = sij - rl;            // log p_i2t[i][j]
        float lp2 = sij - colLSE[j];     // log p_t2i[j][i]
        bool same = (lab[j] == li);
        float q1  = same ? qsi    : qdi;     // q[i][j]
        float lq1 = same ? lqsi   : lqdi;
        float q2  = same ? qs[j]  : qd[j];   // q[j][i]
        float lq2 = same ? lqs[j] : lqd[j];
        acc += q1 * (lq1 - lp1) + expf(lp1) * (lp1 - lq1)
             + q2 * (lq2 - lp2) + expf(lp2) * (lp2 - lq2);
    }
    red[threadIdx.x] = acc; __syncthreads();
    for (int off = 128; off > 0; off >>= 1) {
        if (threadIdx.x < off) red[threadIdx.x] += red[threadIdx.x + off];
        __syncthreads();
    }
    if (threadIdx.x == 0) partials[blockIdx.x] = red[0];
}

// ---------------- 7. final deterministic reduction ----------------
__global__ void kfinal(const float* __restrict__ partials, float* __restrict__ out) {
    __shared__ float red[256];
    float s = 0.f;
    for (int i = threadIdx.x; i < 4096; i += 256) s += partials[i];
    red[threadIdx.x] = s; __syncthreads();
    for (int off = 128; off > 0; off >>= 1) {
        if (threadIdx.x < off) red[threadIdx.x] += red[threadIdx.x + off];
        __syncthreads();
    }
    if (threadIdx.x == 0) out[0] = red[0] * (1.0f / (4.0f * (float)BS));
}

extern "C" void kernel_launch(void* const* d_in, const int* in_sizes, int n_in,
                              void* d_out, int out_size, void* d_ws, size_t ws_size,
                              hipStream_t stream) {
    const float* img   = (const float*)d_in[0];
    const float* tab   = (const float*)d_in[1];
    const float* scale = (const float*)d_in[2];
    const int*   lab   = (const int*)d_in[3];

    // Workspace layout (~80.2 MB total)
    char* ws = (char*)d_ws;
    float*    S       = (float*)ws;                              // 64 MB
    _Float16* img_h   = (_Float16*)(ws + ((size_t)64 << 20));    // 8 MB
    _Float16* tab_h   = (_Float16*)(ws + ((size_t)72 << 20));    // 8 MB
    float*    inv_img = (float*)(ws + ((size_t)80 << 20));       // 4096 each below
    float*    inv_tab = inv_img + BS;
    float*    rowLSE  = inv_tab + BS;
    float*    colLSE  = rowLSE + BS;
    float*    qs      = colLSE + BS;
    float*    qd      = qs + BS;
    float*    lqs     = qd + BS;
    float*    lqd     = lqs + BS;
    float*    parts   = lqd + BS;                                // 4096
    float*    n1      = parts + 4096;                            // 1

    knorm   <<<2 * BS, 256, 0, stream>>>(img, tab, inv_img, inv_tab);
    kconvert<<<8192, 256, 0, stream>>>(img, tab, inv_img, inv_tab, img_h, tab_h);
    kcount  <<<1, 256, 0, stream>>>(lab, n1);
    kqprep  <<<BS / 256, 256, 0, stream>>>(lab, n1, qs, qd, lqs, lqd);
    dim3 gg(BS / 128, BS / 64);
    kgemm   <<<gg, 256, 0, stream>>>(img_h, tab_h, scale, S);
    krowlse <<<BS, 256, 0, stream>>>(S, rowLSE);
    kcollse <<<BS / 64, 256, 0, stream>>>(S, colLSE);
    kreduce <<<4096, 256, 0, stream>>>(S, rowLSE, colLSE, lab, qs, qd, lqs, lqd, parts);
    kfinal  <<<1, 256, 0, stream>>>(parts, (float*)d_out);
}
